// CustomMultiHeadAttention_22067541967809
// MI455X (gfx1250) — compile-verified, matrix operand order checked by reading, not run on hardware
//
#include <hip/hip_runtime.h>
#include <math.h>

// ---------------------------------------------------------------------------
// MHA forward for MI455X (gfx1250): bf16 WMMA (V_WMMA_F32_16X16X32_BF16),
// f32 accumulate, wave32, LDS staging via async global->LDS where available.
// ---------------------------------------------------------------------------

typedef __bf16 bf16_t;
typedef __attribute__((ext_vector_type(16))) __bf16 v16bf;
typedef __attribute__((ext_vector_type(8)))  __bf16 v8bf;
typedef __attribute__((ext_vector_type(8)))  float  v8f;
typedef __attribute__((ext_vector_type(4)))  int    v4i;

#define D_MODEL 1024
#define N_HEADS 16
#define D_K     64
#define BATCH   4
#define SEQ     2048
#define MROWS   (BATCH * SEQ)        // 8192

#if __has_builtin(__builtin_amdgcn_global_load_async_to_lds_b128)
#define HAVE_ASYNC_LDS 1
typedef __attribute__((address_space(1))) v4i gv4i;   // global
typedef __attribute__((address_space(3))) v4i lv4i;   // LDS
#endif

__device__ __forceinline__ void async_copy_b128(const void* g, void* l) {
#ifdef HAVE_ASYNC_LDS
    __builtin_amdgcn_global_load_async_to_lds_b128((gv4i*)g, (lv4i*)l, 0, 0);
#else
    *(v8bf*)l = *(const v8bf*)g;     // sync fallback: global load + ds_store
#endif
}

__device__ __forceinline__ void stage_wait() {
#ifdef HAVE_ASYNC_LDS
#if __has_builtin(__builtin_amdgcn_s_wait_asynccnt)
    __builtin_amdgcn_s_wait_asynccnt(0);
#else
    asm volatile("s_wait_asynccnt 0" ::: "memory");
#endif
#endif
    __syncthreads();
}

__device__ __forceinline__ v8f wmma_bf16(v16bf a, v16bf b, v8f c) {
    // (neg_a, A, neg_b, B, c_mod, C, reuse_a, reuse_b)
    return __builtin_amdgcn_wmma_f32_16x16x32_bf16(false, a, false, b,
                                                   (short)0, c, false, false);
}

// Load a 16x32 bf16 A/B fragment from a row-major [16][K] tile (global or LDS).
// ISA 7.12.2: lane<16 -> row=lane, k = {0..7, 16..23}; lane>=16 -> row=lane-16,
// k = {8..15, 24..31}. Two contiguous 16-byte loads per lane.
__device__ __forceinline__ v16bf load_frag_b16(const bf16_t* tile, int ld, int lane) {
    const bf16_t* p = tile + (size_t)(lane & 15) * ld + ((lane >> 4) * 8);
    union { v16bf v; v8bf h[2]; } u;
    u.h[0] = *(const v8bf*)(p);
    u.h[1] = *(const v8bf*)(p + 16);
    return u.v;
}

// ---------------------------------------------------------------------------
// 1) f32 -> bf16 convert
// ---------------------------------------------------------------------------
__global__ void cvt_f32_bf16_kernel(const float* __restrict__ src,
                                    bf16_t* __restrict__ dst, int n) {
    int i = blockIdx.x * blockDim.x + threadIdx.x;
    if (i < n) dst[i] = (bf16_t)src[i];
}

// ---------------------------------------------------------------------------
// 2) GEMM: C[M=8192, N=1024] = A(bf16,[M][1024]) @ W(bf16,[N][1024])^T + bias
//    Block = 8 waves: one 16x1024 A strip staged in LDS (32KB, async copy),
//    each wave owns one of 8 consecutive 16x16 N-tiles.
//    mode 0: out bf16 at [b,h,s,dk]   (Q / K projections)
//    mode 1: out bf16 at [b,h,dk,s]   (V projection, transposed)
//    mode 2: out f32  at [m][n]       (final output projection)
// ---------------------------------------------------------------------------
__global__ void gemm_bf16_kernel(const bf16_t* __restrict__ A,
                                 const bf16_t* __restrict__ W,
                                 const float*  __restrict__ bias,
                                 bf16_t* __restrict__ outb,
                                 float*  __restrict__ outf,
                                 int mode) {
    __shared__ __align__(16) bf16_t As[16 * D_MODEL];   // 32 KB
    const int t    = threadIdx.x;
    const int lane = t & 31;
    const int wave = t >> 5;
    const int mt     = blockIdx.x >> 3;          // 512 M-tiles
    const int ntBase = (blockIdx.x & 7) * 8;     // 8 groups of 8 N-tiles

    // Stage A strip: 2048 chunks of 16B, 8 per thread.
    const bf16_t* Ab = A + (size_t)mt * 16 * D_MODEL;
#pragma unroll
    for (int j = 0; j < 8; ++j) {
        const int c = t + 256 * j;
        async_copy_b128(Ab + c * 8, As + c * 8);
    }
    stage_wait();

    const int nt = ntBase + wave;
    const bf16_t* Wb = W + (size_t)nt * 16 * D_MODEL;

    v8f c = {};
#pragma unroll 4
    for (int kk = 0; kk < D_MODEL; kk += 32) {
        v16bf af = load_frag_b16(As + kk, D_MODEL, lane);   // ds_load_b128 x2
        v16bf bf = load_frag_b16(Wb + kk, D_MODEL, lane);   // global_load_b128 x2
        c = wmma_bf16(af, bf, c);
    }

    const int n     = nt * 16 + (lane & 15);
    const int mBase = mt * 16 + (lane >> 4) * 8;
    const float bv  = bias[n];
    const int h = n >> 6, dk = n & 63;
#pragma unroll
    for (int r = 0; r < 8; ++r) {
        const int m = mBase + r;
        const float v = c[r] + bv;
        if (mode == 2) {
            outf[(size_t)m * D_MODEL + n] = v;
        } else {
            const int b = m >> 11, s = m & 2047;
            const size_t idx = (mode == 0)
                ? (((size_t)(b * N_HEADS + h) * SEQ + s) * D_K + dk)
                : (((size_t)(b * N_HEADS + h) * D_K + dk) * SEQ + s);
            outb[idx] = (bf16_t)v;
        }
    }
}

// ---------------------------------------------------------------------------
// 3) scores[b,h,sq,sk] = (Q . K) * 8.0  (ref: /(1/sqrt(dk)));  mask==0 -> -1e9
//    Each wave keeps its Q fragments in registers and sweeps 16 K-tiles.
// ---------------------------------------------------------------------------
__global__ void scores_kernel(const bf16_t* __restrict__ Qp,
                              const bf16_t* __restrict__ Kp,
                              const int*    __restrict__ mask,
                              float* __restrict__ attn) {
    const int lane = threadIdx.x & 31;
    const long wid = (long)blockIdx.x * 8 + (threadIdx.x >> 5); // 2^16 wave jobs
    const int kg = (int)(wid & 7);            // group of 16 K-tiles
    const int qt = (int)((wid >> 3) & 127);
    const int h  = (int)((wid >> 10) & 15);
    const int b  = (int)(wid >> 14);

    const size_t headBase = (size_t)(b * N_HEADS + h) * SEQ;
    const bf16_t* Qb = Qp + (headBase + qt * 16) * D_K;
    const v16bf a0 = load_frag_b16(Qb + 0,  D_K, lane);
    const v16bf a1 = load_frag_b16(Qb + 32, D_K, lane);

    const int sqBase = qt * 16 + (lane >> 4) * 8;
    for (int kt = kg * 16; kt < kg * 16 + 16; ++kt) {
        const bf16_t* Kb = Kp + (headBase + kt * 16) * D_K;
        v16bf b0 = load_frag_b16(Kb + 0,  D_K, lane);
        v16bf b1 = load_frag_b16(Kb + 32, D_K, lane);
        v8f c = {};
        c = wmma_bf16(a0, b0, c);
        c = wmma_bf16(a1, b1, c);

        const int sk = kt * 16 + (lane & 15);
        float* orow = attn + (headBase + sqBase) * SEQ + sk;
#pragma unroll
        for (int r = 0; r < 8; ++r) {
            float v = c[r] * 8.0f;
            if (mask[(size_t)(sqBase + r) * SEQ + sk] == 0) v = -1e9f;
            orow[(size_t)r * SEQ] = v;
        }
    }
}

// ---------------------------------------------------------------------------
// 4) In-place row softmax over attn rows of length SEQ (one block per row).
// ---------------------------------------------------------------------------
__global__ void softmax_rows_kernel(float* __restrict__ attn) {
    __shared__ float red[256];
    float* row = attn + (size_t)blockIdx.x * SEQ;
    const int t = threadIdx.x;

    float mx = -INFINITY;
    for (int i = t; i < SEQ; i += 256) mx = fmaxf(mx, row[i]);
    red[t] = mx; __syncthreads();
    for (int s = 128; s > 0; s >>= 1) {
        if (t < s) red[t] = fmaxf(red[t], red[t + s]);
        __syncthreads();
    }
    mx = red[0]; __syncthreads();

    float sum = 0.0f;
    for (int i = t; i < SEQ; i += 256) sum += __expf(row[i] - mx);
    red[t] = sum; __syncthreads();
    for (int s = 128; s > 0; s >>= 1) {
        if (t < s) red[t] += red[t + s];
        __syncthreads();
    }
    const float inv = 1.0f / red[0];

    for (int i = t; i < SEQ; i += 256) row[i] = __expf(row[i] - mx) * inv;
}

// ---------------------------------------------------------------------------
// 5) context: one block (4 waves, 128 thr) per (b,h,q-tile). The 16xSEQ attn
//    strip is converted f32->bf16 into 64KB LDS once; each wave computes one
//    16x16 dk-tile:  attn[16xSEQ] @ V[SEQ x 16]  (V held transposed).
//    ctx written bf16 as [B*S, D_MODEL] for the final GEMM.
// ---------------------------------------------------------------------------
__global__ void context_kernel(const float*  __restrict__ attn,
                               const bf16_t* __restrict__ Vt,
                               bf16_t* __restrict__ ctx) {
    __shared__ __align__(16) bf16_t As[16 * SEQ];       // 64 KB
    const int t    = threadIdx.x;   // 128 threads
    const int lane = t & 31;
    const int wave = t >> 5;        // nt = wave (0..3)
    const int qt = blockIdx.x & 127;
    const int h  = (blockIdx.x >> 7) & 15;
    const int b  = blockIdx.x >> 11;

    const float* Ab = attn + ((size_t)(b * N_HEADS + h) * SEQ + qt * 16) * SEQ;
    // Convert strip into LDS: 8192 float4 chunks, 64 per thread.
#pragma unroll 8
    for (int j = 0; j < 64; ++j) {
        const int cidx = t + 128 * j;
        const float4 f = ((const float4*)Ab)[cidx];
        bf16_t* d = As + cidx * 4;
        d[0] = (bf16_t)f.x; d[1] = (bf16_t)f.y;
        d[2] = (bf16_t)f.z; d[3] = (bf16_t)f.w;
    }
    __syncthreads();

    const bf16_t* Bb = Vt + ((size_t)(b * N_HEADS + h) * D_K + wave * 16) * SEQ;
    v8f c = {};
#pragma unroll 4
    for (int kk = 0; kk < SEQ; kk += 32) {
        v16bf af = load_frag_b16(As + kk, SEQ, lane);   // ds_load_b128 x2
        v16bf bf = load_frag_b16(Bb + kk, SEQ, lane);   // global_load_b128 x2
        c = wmma_bf16(af, bf, c);
    }

    const int dk    = wave * 16 + (lane & 15);
    const int sBase = qt * 16 + (lane >> 4) * 8;
#pragma unroll
    for (int r = 0; r < 8; ++r) {
        ctx[((size_t)b * SEQ + sBase + r) * D_MODEL + h * D_K + dk] = (bf16_t)c[r];
    }
}

// ---------------------------------------------------------------------------
// Host launcher
// ---------------------------------------------------------------------------
extern "C" void kernel_launch(void* const* d_in, const int* in_sizes, int n_in,
                              void* d_out, int out_size, void* d_ws, size_t ws_size,
                              hipStream_t stream) {
    (void)in_sizes; (void)n_in; (void)out_size; (void)ws_size;

    const float* query = (const float*)d_in[0];
    const float* key   = (const float*)d_in[1];
    const float* value = (const float*)d_in[2];
    const float* wq    = (const float*)d_in[3];
    const float* bq    = (const float*)d_in[4];
    const float* wk    = (const float*)d_in[5];
    const float* bk    = (const float*)d_in[6];
    const float* wv    = (const float*)d_in[7];
    const float* bv    = (const float*)d_in[8];
    const float* wo    = (const float*)d_in[9];
    const float* bo    = (const float*)d_in[10];
    const int*   mask  = (const int*)d_in[11];

    float* out  = (float*)d_out;                       // [B,S,D] = 8,388,608
    float* attn = out + (size_t)MROWS * D_MODEL;       // [B,H,S,S]

    const size_t MD = (size_t)MROWS * D_MODEL;   // 8,388,608 elems
    const size_t DD = (size_t)D_MODEL * D_MODEL; // 1,048,576 elems

    bf16_t* p   = (bf16_t*)d_ws;
    bf16_t* xq  = p; p += MD;
    bf16_t* xk  = p; p += MD;
    bf16_t* xv  = p; p += MD;
    bf16_t* wqb = p; p += DD;
    bf16_t* wkb = p; p += DD;
    bf16_t* wvb = p; p += DD;
    bf16_t* wob = p; p += DD;
    bf16_t* Qp  = p; p += MD;   // [b,h,s,dk]
    bf16_t* Kp  = p; p += MD;   // [b,h,s,dk]
    bf16_t* Vt  = p; p += MD;   // [b,h,dk,s]
    bf16_t* ctx = p; p += MD;   // [B*S, D]

    const int cvtBig   = (int)((MD + 255) / 256);
    const int cvtSmall = (int)((DD + 255) / 256);

    // 1) converts
    cvt_f32_bf16_kernel<<<cvtBig,   256, 0, stream>>>(query, xq,  (int)MD);
    cvt_f32_bf16_kernel<<<cvtBig,   256, 0, stream>>>(key,   xk,  (int)MD);
    cvt_f32_bf16_kernel<<<cvtBig,   256, 0, stream>>>(value, xv,  (int)MD);
    cvt_f32_bf16_kernel<<<cvtSmall, 256, 0, stream>>>(wq,    wqb, (int)DD);
    cvt_f32_bf16_kernel<<<cvtSmall, 256, 0, stream>>>(wk,    wkb, (int)DD);
    cvt_f32_bf16_kernel<<<cvtSmall, 256, 0, stream>>>(wv,    wvb, (int)DD);
    cvt_f32_bf16_kernel<<<cvtSmall, 256, 0, stream>>>(wo,    wob, (int)DD);

    // 2) projections: 512 M-tiles x 8 N-groups = 4096 blocks
    gemm_bf16_kernel<<<4096, 256, 0, stream>>>(xq, wqb, bq, Qp, nullptr, 0);
    gemm_bf16_kernel<<<4096, 256, 0, stream>>>(xk, wkb, bk, Kp, nullptr, 0);
    gemm_bf16_kernel<<<4096, 256, 0, stream>>>(xv, wvb, bv, Vt, nullptr, 1);

    // 3) scores: 2^16 wave-jobs / 8 waves = 8192 blocks
    scores_kernel<<<8192, 256, 0, stream>>>(Qp, Kp, mask, attn);

    // 4) softmax: one block per row, B*H*S = 131072 rows
    softmax_rows_kernel<<<131072, 256, 0, stream>>>(attn);

    // 5) context: one block per (b,h,qt) = 8192 blocks of 128 threads
    context_kernel<<<8192, 128, 0, stream>>>(attn, Vt, ctx);

    // 6) output projection (f32 + bias) into d_out
    gemm_bf16_kernel<<<4096, 256, 0, stream>>>(ctx, wob, bo, nullptr, out, 2);
}
